// RNNDecoder_80341658239145
// MI455X (gfx1250) — matrix operand order repaired and checked by reading, not compiled, not run on hardware
//
#include <hip/hip_runtime.h>
#include <hip/hip_bf16.h>
#include <math.h>
#include <stdint.h>

typedef __bf16 bf16_t;
typedef bf16_t v16bf __attribute__((ext_vector_type(16)));
typedef bf16_t v8bf  __attribute__((ext_vector_type(8)));
typedef bf16_t v4bf  __attribute__((ext_vector_type(4)));
typedef float  v8f   __attribute__((ext_vector_type(8)));

// ---------------------------------------------------------------------------
// WMMA fragment helpers (wave32, v_wmma_f32_16x16x32_bf16)
// A (16x32, MxK): lane&15 = M row, lane>>4 selects K-halves {0..7,16..23} vs {8..15,24..31}
// B (32x16, KxN): lane&15 = N col, lane>>4 selects the same K-halves as A
// C (16x16 f32):  lanes<16: m=vgpr,n=lane ; lanes>=16: m=vgpr+8,n=lane-16
// ---------------------------------------------------------------------------
static __device__ __forceinline__ v16bf frag_a_from_lds(const bf16_t* base, int str, int lane) {
    int r  = lane & 15;
    int hl = lane >> 4;
    const bf16_t* p = base + r * str;
    v8bf lo = *(const v8bf*)(p + hl * 8);
    v8bf hi = *(const v8bf*)(p + 16 + hl * 8);
    v16bf f;
#pragma unroll
    for (int i = 0; i < 8; ++i) { f[i] = lo[i]; f[i + 8] = hi[i]; }
    return f;
}

static __device__ __forceinline__ v16bf frag_b_from_lds(const bf16_t* base, int str, int lane) {
    const int hl = lane >> 4;
    const bf16_t* p = base + (lane & 15);   // lane&15 = n column
    v16bf f;
#pragma unroll
    for (int i = 0; i < 8; ++i) { f[i] = p[(hl * 8 + i) * str]; f[i + 8] = p[(16 + hl * 8 + i) * str]; }
    return f;
}

// 16-byte async copy global -> LDS (CDNA5 ASYNCcnt path)
static __device__ __forceinline__ void async_copy16(unsigned lds_off, const bf16_t* gptr) {
    asm volatile("global_load_async_to_lds_b128 %0, %1, off"
                 :: "v"(lds_off), "v"((unsigned long long)(uintptr_t)gptr)
                 : "memory");
}

// ---------------------------------------------------------------------------
// Tiled GEMM on pre-converted, zero-padded bf16 operands:
//   C[M,N](fp32) = act( A[M,Kp] @ B[Kp,Np] (+ C) + bias )
// M, grid are exact multiples of 128; Kp multiple of 32; B columns padded to
// gridDim.x*128. No load guards -> tiles staged with async b128 copies,
// double-buffered (steady state: s_wait_asynccnt 4).
// ---------------------------------------------------------------------------
#define BM 128
#define BN 128
#define BK 32
#define ASTR 40    // padded LDS row stride (elems) for A tile [m][k]
#define BSTR 136   // padded LDS row stride (elems) for B tile [k][n]

__global__ __launch_bounds__(256) void gemm_wmma_async(
    const bf16_t* __restrict__ A, long long sA, int lda,
    const bf16_t* __restrict__ B, long long sB, int ldb,
    float* __restrict__ C, long long sC, int ldc,
    int N, int K,
    const float* __restrict__ bias, int accum, int act_tanh)
{
    __shared__ bf16_t As[2][BM * ASTR];
    __shared__ bf16_t Bs[2][BK * BSTR];

    A += (long long)blockIdx.z * sA;
    B += (long long)blockIdx.z * sB;
    C += (long long)blockIdx.z * sC;

    const int m0 = blockIdx.y * BM;
    const int n0 = blockIdx.x * BN;
    const int tid  = threadIdx.x;
    const int lane = tid & 31;
    const int wave = tid >> 5;
    const int wm = wave >> 1;  // 0..3 -> 32-row slab
    const int wn = wave & 1;   // 0..1 -> 64-col slab

    // per-thread 16B staging chunks (2 for A, 2 for B => 4 async issues / wave / tile)
    const int a_r0 = (tid)       >> 2, a_c0 = ((tid)       & 3) << 3;
    const int a_r1 = (tid + 256) >> 2, a_c1 = ((tid + 256) & 3) << 3;
    const int b_k0 = (tid)       >> 4, b_n0 = ((tid)       & 15) << 3;
    const int b_k1 = (tid + 256) >> 4, b_n1 = ((tid + 256) & 15) << 3;

    v8f acc[2][4];
#pragma unroll
    for (int i = 0; i < 2; ++i)
#pragma unroll
        for (int j = 0; j < 4; ++j)
#pragma unroll
            for (int e = 0; e < 8; ++e) acc[i][j][e] = 0.0f;

    const int nk = K / BK;

    // stage tile 0
    {
        async_copy16((unsigned)(uintptr_t)&As[0][a_r0 * ASTR + a_c0],
                     A + (long long)(m0 + a_r0) * lda + a_c0);
        async_copy16((unsigned)(uintptr_t)&As[0][a_r1 * ASTR + a_c1],
                     A + (long long)(m0 + a_r1) * lda + a_c1);
        async_copy16((unsigned)(uintptr_t)&Bs[0][b_k0 * BSTR + b_n0],
                     B + (long long)b_k0 * ldb + n0 + b_n0);
        async_copy16((unsigned)(uintptr_t)&Bs[0][b_k1 * BSTR + b_n1],
                     B + (long long)b_k1 * ldb + n0 + b_n1);
    }

    for (int kt = 0; kt < nk; ++kt) {
        if (kt + 1 < nk) {
            const int kn  = (kt + 1) * BK;
            const int buf = (kt + 1) & 1;
            async_copy16((unsigned)(uintptr_t)&As[buf][a_r0 * ASTR + a_c0],
                         A + (long long)(m0 + a_r0) * lda + kn + a_c0);
            async_copy16((unsigned)(uintptr_t)&As[buf][a_r1 * ASTR + a_c1],
                         A + (long long)(m0 + a_r1) * lda + kn + a_c1);
            async_copy16((unsigned)(uintptr_t)&Bs[buf][b_k0 * BSTR + b_n0],
                         B + (long long)(kn + b_k0) * ldb + n0 + b_n0);
            async_copy16((unsigned)(uintptr_t)&Bs[buf][b_k1 * BSTR + b_n1],
                         B + (long long)(kn + b_k1) * ldb + n0 + b_n1);
            // async loads complete in order: cnt<=4 => current tile has landed
            asm volatile("s_wait_asynccnt 0x4" ::: "memory");
        } else {
            asm volatile("s_wait_asynccnt 0x0" ::: "memory");
        }
        __syncthreads();

        const bf16_t* Ab = As[kt & 1];
        const bf16_t* Bb = Bs[kt & 1];

        v16bf af[2], bfr[4];
#pragma unroll
        for (int i = 0; i < 2; ++i)
            af[i] = frag_a_from_lds(Ab + (wm * 2 + i) * 16 * ASTR, ASTR, lane);
#pragma unroll
        for (int j = 0; j < 4; ++j)
            bfr[j] = frag_b_from_lds(Bb + (wn * 4 + j) * 16, BSTR, lane);
#pragma unroll
        for (int i = 0; i < 2; ++i)
#pragma unroll
            for (int j = 0; j < 4; ++j)
                acc[i][j] = __builtin_amdgcn_wmma_f32_16x16x32_bf16(
                    false, af[i], false, bfr[j], (short)0, acc[i][j], false, false);
        __syncthreads();
    }

    // ---- epilogue (store guard only on N; M/grid are exact) ----
    const int hl = lane >> 4, nl = lane & 15;
#pragma unroll
    for (int i = 0; i < 2; ++i) {
#pragma unroll
        for (int j = 0; j < 4; ++j) {
            const int gn = n0 + (wn * 4 + j) * 16 + nl;
            if (gn < N) {
#pragma unroll
                for (int vg = 0; vg < 8; ++vg) {
                    const int gm = m0 + (wm * 2 + i) * 16 + vg + hl * 8;
                    long long o = (long long)gm * ldc + gn;
                    float v = acc[i][j][vg];
                    if (accum)    v += C[o];
                    if (bias)     v += bias[gn];
                    if (act_tanh) v = tanhf(v);
                    C[o] = v;
                }
            }
        }
    }
}

// ---------------------------------------------------------------------------
// fp32 -> zero-padded bf16 converters (row-major, optional batch; and a
// batched transpose variant used to materialize enc^T once).
// ---------------------------------------------------------------------------
__global__ __launch_bounds__(256) void cvt_pad_bf16(
    const float* __restrict__ src, long long sS, int lds_,
    bf16_t* __restrict__ dst, long long sD, int ldd,
    int R, int C)
{
    const int c = blockIdx.x * 256 + threadIdx.x;
    const int r = blockIdx.y;
    if (c >= ldd) return;
    float v = 0.0f;
    if (r < R && c < C) v = src[(long long)blockIdx.z * sS + (long long)r * lds_ + c];
    dst[(long long)blockIdx.z * sD + (long long)r * ldd + c] = (bf16_t)v;
}

__global__ __launch_bounds__(256) void cvt_padT_bf16(
    const float* __restrict__ src, long long sS, int lds_,   // src [R][lds_]
    bf16_t* __restrict__ dst, long long sD, int ldd,         // dst [C][ldd]; dst[c][r]=src[r][c]
    int R, int C)
{
    const int r = blockIdx.x * 256 + threadIdx.x;  // dst column (source row)
    const int c = blockIdx.y;                      // dst row (source column)
    if (r >= ldd) return;
    float v = 0.0f;
    if (r < R && c < C) v = src[(long long)blockIdx.z * sS + (long long)r * lds_ + c];
    dst[(long long)blockIdx.z * sD + (long long)c * ldd + r] = (bf16_t)v;
}

// ---------------------------------------------------------------------------
// Persistent RNN scan: 16 WGs x 64 threads. WG g owns output columns
// [g*32, g*32+32) with those Wh columns resident in LDS as bf16.
// h_t = tanh(xproj[:,t,:] + h_{t-1} @ Wh + b); M=8 state fits one 16-row tile.
// Steps sync via monotonic device-scope atomic barrier (release add /
// acquire spin + s_sleep + agent acquire fence).
// ---------------------------------------------------------------------------
#define SCAN_WGS 16
#define SCAN_COLS 32
#define WHSTR 40

__global__ void init_ctr_kernel(unsigned int* ctr) { *ctr = 0u; }

__global__ __launch_bounds__(64) void rnn_scan_kernel(
    const float* __restrict__ xproj,    // [8,512,512]
    const float* __restrict__ h0,       // [8,512]
    const float* __restrict__ Wh,       // [512,512]
    const float* __restrict__ bvec,     // [512]
    float* __restrict__ outputs,        // [8,512,512]
    unsigned int* __restrict__ ctr)
{
    const int H = 512, T = 512;
    __shared__ bf16_t WhS[512 * WHSTR];   // [k][c_local], 40,960 B

    const int c0   = blockIdx.x * SCAN_COLS;
    const int tid  = threadIdx.x;
    const int lane = tid & 31;
    const int wave = tid >> 5;            // 0..1, each owns one 16-col subtile

    for (int fid = tid; fid < 512 * 8; fid += 64) {
        int k  = fid >> 3;
        int cq = (fid & 7) << 2;
        const float4 v = *(const float4*)(Wh + (long long)k * H + c0 + cq);
        v4bf w; w[0] = (bf16_t)v.x; w[1] = (bf16_t)v.y; w[2] = (bf16_t)v.z; w[3] = (bf16_t)v.w;
        *(v4bf*)(&WhS[k * WHSTR + cq]) = w;
    }
    __syncthreads();

    const int r  = lane & 15;
    const int hl = lane >> 4;
    const bool rvalid = (r < 8);

    for (int t = 0; t < T; ++t) {
        const float* hbase   = (t == 0) ? h0 : (outputs + (long long)(t - 1) * H);
        const long long rstr = (t == 0) ? (long long)H : (long long)T * H;

        v8f acc;
#pragma unroll
        for (int e = 0; e < 8; ++e) acc[e] = 0.0f;

        const float* rowp = hbase + (long long)r * rstr;
#pragma unroll 4
        for (int k0 = 0; k0 < 512; k0 += 32) {
            v16bf a;
            if (rvalid) {
                const float* p = rowp + k0 + hl * 8;
#pragma unroll
                for (int i = 0; i < 8; ++i) { a[i] = (bf16_t)p[i]; a[i + 8] = (bf16_t)p[16 + i]; }
            } else {
#pragma unroll
                for (int i = 0; i < 16; ++i) a[i] = (bf16_t)0.0f;
            }
            v16bf b = frag_b_from_lds(&WhS[k0 * WHSTR + wave * 16], WHSTR, lane);
            acc = __builtin_amdgcn_wmma_f32_16x16x32_bf16(
                false, a, false, b, (short)0, acc, false, false);
        }

        if (hl == 0) {
            int col = c0 + wave * 16 + (lane & 15);
#pragma unroll
            for (int n = 0; n < 8; ++n) {
                long long o = (long long)n * T * H + (long long)t * H + col;
                outputs[o] = tanhf(acc[n] + xproj[o] + bvec[col]);
            }
        }

        __threadfence();
        __syncthreads();
        if (tid == 0) {
            __hip_atomic_fetch_add(ctr, 1u, __ATOMIC_RELEASE, __HIP_MEMORY_SCOPE_AGENT);
            const unsigned target = (unsigned)(SCAN_WGS * (t + 1));
            while (__hip_atomic_load(ctr, __ATOMIC_ACQUIRE, __HIP_MEMORY_SCOPE_AGENT) < target)
                __builtin_amdgcn_s_sleep(2);
        }
        __syncthreads();
        __builtin_amdgcn_fence(__ATOMIC_ACQUIRE, "agent");
    }
}

// ---------------------------------------------------------------------------
__global__ __launch_bounds__(256) void copy_final_kernel(
    const float* __restrict__ outputs, float* __restrict__ dst)
{
    int i = blockIdx.x * blockDim.x + threadIdx.x;
    if (i < 8 * 512) {
        int n = i >> 9, c = i & 511;
        dst[i] = outputs[(long long)n * 512 * 512 + 511 * 512 + c];
    }
}

__global__ __launch_bounds__(256) void softmax_rows_kernel(
    const float* __restrict__ S, float* __restrict__ P, int L)
{
    const long long row = blockIdx.x;
    const float* s = S + row * L;
    float*       p = P + row * L;
    __shared__ float red[8];
    const int tid = threadIdx.x, lane = tid & 31, wv = tid >> 5;

    float m = -3.0e38f;
    for (int i = tid; i < L; i += 256) m = fmaxf(m, s[i]);
#pragma unroll
    for (int o = 16; o > 0; o >>= 1) m = fmaxf(m, __shfl_xor(m, o, 32));
    if (lane == 0) red[wv] = m;
    __syncthreads();
    if (tid == 0) {
        float mm = red[0];
        for (int i = 1; i < 8; ++i) mm = fmaxf(mm, red[i]);
        red[0] = mm;
    }
    __syncthreads();
    m = red[0];
    __syncthreads();

    float sum = 0.0f;
    for (int i = tid; i < L; i += 256) { float e = __expf(s[i] - m); p[i] = e; sum += e; }
#pragma unroll
    for (int o = 16; o > 0; o >>= 1) sum += __shfl_xor(sum, o, 32);
    if (lane == 0) red[wv] = sum;
    __syncthreads();
    if (tid == 0) {
        float ss = 0.0f;
        for (int i = 0; i < 8; ++i) ss += red[i];
        red[0] = ss;
    }
    __syncthreads();
    const float inv = 1.0f / red[0];
    for (int i = tid; i < L; i += 256) p[i] *= inv;
}

// ---------------------------------------------------------------------------
extern "C" void kernel_launch(void* const* d_in, const int* in_sizes, int n_in,
                              void* d_out, int out_size, void* d_ws, size_t ws_size,
                              hipStream_t stream)
{
    (void)in_sizes; (void)n_in; (void)out_size; (void)ws_size;

    const float* emb = (const float*)d_in[0];   // (8,512,512)
    const float* h0  = (const float*)d_in[1];   // (8,1,512)
    const float* enc = (const float*)d_in[2];   // (8,196,512)
    const float* Wx  = (const float*)d_in[3];   // (512,512)
    const float* Wh  = (const float*)d_in[4];   // (512,512)
    const float* bv  = (const float*)d_in[5];   // (512,)
    const float* Wo  = (const float*)d_in[6];   // (512,32000)
    const float* bo  = (const float*)d_in[7];   // (32000,)
    const float* Wa  = (const float*)d_in[8];   // (512,512)
    const float* Wc  = (const float*)d_in[9];   // (1024,512)

    const int Nb = 8, T = 512, H = 512, Cc = 512, L = 196, V = 32000;
    const int MT = Nb * T;          // 4096
    const int Lk = 224;             // L padded to mult of 32 (K side)
    const int Ln = 256;             // L padded to mult of 128 (N side)

    float* out     = (float*)d_out;
    float* logits  = out;                                   // 8*512*32000
    float* final_h = out + (long long)MT * V;               // 8*512
    float* attn    = final_h + (long long)Nb * H;           // 8*512*196

    // ---- fp32 workspace ----
    float* ws      = (float*)d_ws;
    float* xproj   = ws;                                    // 4096*512
    float* outputs = xproj + (long long)MT * H;             // 4096*512
    float* inner   = outputs + (long long)MT * H;           // 4096*512
    float* sbuf    = inner + (long long)MT * Cc;            // 8*512*196
    unsigned int* ctr = (unsigned int*)(sbuf + (long long)MT * L);
    float* context = xproj;                                 // reuse (scan done)
    float* att_out = inner;                                 // reuse (inner_bf done)
    float* fp_end  = sbuf + (long long)MT * L + 16;

    // ---- padded bf16 workspace ----
    bf16_t* bp = (bf16_t*)fp_end;
    bf16_t* emb_bf   = bp;  bp += (long long)MT * H;        // [4096][512]
    bf16_t* Wx_bf    = bp;  bp += (long long)H * H;         // [512][512]
    bf16_t* Wa_bf    = bp;  bp += (long long)H * Cc;        // [512][512]
    bf16_t* Wc_bf    = bp;  bp += (long long)(Cc + H) * H;  // [1024][512]
    bf16_t* Wo_bf    = bp;  bp += (long long)H * V;         // [512][32000]
    bf16_t* enc_bf   = bp;  bp += (long long)Nb * Lk * Cc;  // [8][224][512]
    bf16_t* encT_bf  = bp;  bp += (long long)Nb * Cc * Ln;  // [8][512][256]
    bf16_t* outs_bf  = bp;  bp += (long long)MT * H;        // [4096][512]
    bf16_t* inner_bf = bp;  bp += (long long)MT * Cc;       // [4096][512]
    bf16_t* attn_bf  = bp;  bp += (long long)MT * Lk;       // [4096][224]
    bf16_t* ctx_bf   = bp;  bp += (long long)MT * Cc;       // [4096][512]
    bf16_t* aout_bf  = bp;                                  // [4096][512]

    const dim3 blk(256);

    // ---- convert static operands to padded bf16 ----
    cvt_pad_bf16<<<dim3(2, MT, 1), blk, 0, stream>>>(emb, 0, H, emb_bf, 0, H, MT, H);
    cvt_pad_bf16<<<dim3(2, H, 1), blk, 0, stream>>>(Wx, 0, H, Wx_bf, 0, H, H, H);
    cvt_pad_bf16<<<dim3(2, H, 1), blk, 0, stream>>>(Wa, 0, Cc, Wa_bf, 0, Cc, H, Cc);
    cvt_pad_bf16<<<dim3(2, Cc + H, 1), blk, 0, stream>>>(Wc, 0, H, Wc_bf, 0, H, Cc + H, H);
    cvt_pad_bf16<<<dim3(V / 256, H, 1), blk, 0, stream>>>(Wo, 0, V, Wo_bf, 0, V, H, V);
    cvt_pad_bf16<<<dim3(2, Lk, Nb), blk, 0, stream>>>(
        enc, (long long)L * Cc, Cc, enc_bf, (long long)Lk * Cc, Cc, L, Cc);
    cvt_padT_bf16<<<dim3(1, Cc, Nb), blk, 0, stream>>>(
        enc, (long long)L * Cc, Cc, encT_bf, (long long)Cc * Ln, Ln, L, Cc);

    // 1) x_proj = emb @ Wx
    gemm_wmma_async<<<dim3(H / BN, MT / BM, 1), blk, 0, stream>>>(
        emb_bf, 0, H, Wx_bf, 0, H, xproj, 0, H, H, H, nullptr, 0, 0);

    // 2) sequential scan
    init_ctr_kernel<<<1, 1, 0, stream>>>(ctr);
    rnn_scan_kernel<<<dim3(SCAN_WGS), dim3(64), 0, stream>>>(
        xproj, h0, Wh, bv, outputs, ctr);

    // 3) final_h = outputs[:, T-1, :]
    copy_final_kernel<<<dim3(16), blk, 0, stream>>>(outputs, final_h);

    // 4) inner = outputs @ Wa
    cvt_pad_bf16<<<dim3(2, MT, 1), blk, 0, stream>>>(outputs, 0, H, outs_bf, 0, H, MT, H);
    gemm_wmma_async<<<dim3(Cc / BN, MT / BM, 1), blk, 0, stream>>>(
        outs_bf, 0, H, Wa_bf, 0, Cc, inner, 0, Cc, Cc, H, nullptr, 0, 0);

    // 5) s[n] = inner[n] @ enc[n]^T
    cvt_pad_bf16<<<dim3(2, MT, 1), blk, 0, stream>>>(inner, 0, Cc, inner_bf, 0, Cc, MT, Cc);
    gemm_wmma_async<<<dim3(Ln / BN, T / BM, Nb), blk, 0, stream>>>(
        inner_bf, (long long)T * Cc, Cc,
        encT_bf, (long long)Cc * Ln, Ln,
        sbuf, (long long)T * L, L,
        L, Cc, nullptr, 0, 0);

    // 6) attn = softmax(s) over L (straight into d_out)
    softmax_rows_kernel<<<dim3(MT), blk, 0, stream>>>(sbuf, attn, L);

    // 7) context[n] = attn[n] @ enc[n]   (K padded 196 -> 224 with zeros)
    cvt_pad_bf16<<<dim3(1, MT, 1), blk, 0, stream>>>(attn, 0, L, attn_bf, 0, Lk, MT, L);
    gemm_wmma_async<<<dim3(Cc / BN, T / BM, Nb), blk, 0, stream>>>(
        attn_bf, (long long)T * Lk, Lk,
        enc_bf, (long long)Lk * Cc, Cc,
        context, (long long)T * Cc, Cc,
        Cc, Lk, nullptr, 0, 0);

    // 8) att_out = tanh(context @ Wc[0:512] + outputs @ Wc[512:1024])
    cvt_pad_bf16<<<dim3(2, MT, 1), blk, 0, stream>>>(context, 0, Cc, ctx_bf, 0, Cc, MT, Cc);
    gemm_wmma_async<<<dim3(H / BN, MT / BM, 1), blk, 0, stream>>>(
        ctx_bf, 0, Cc, Wc_bf, 0, H, att_out, 0, H, H, Cc, nullptr, 0, 0);
    gemm_wmma_async<<<dim3(H / BN, MT / BM, 1), blk, 0, stream>>>(
        outs_bf, 0, H, Wc_bf + (long long)Cc * H, 0, H, att_out, 0, H, H, H,
        nullptr, 1, 1);

    // 9) logits = att_out @ Wo + bo
    cvt_pad_bf16<<<dim3(2, MT, 1), blk, 0, stream>>>(att_out, 0, H, aout_bf, 0, H, MT, H);
    gemm_wmma_async<<<dim3(V / BN, MT / BM, 1), blk, 0, stream>>>(
        aout_bf, 0, H, Wo_bf, 0, V, logits, 0, V, V, H, bo, 0, 0);
}
